// PerformerAttention_12120397709580
// MI455X (gfx1250) — matrix operand order repaired and checked by reading, not compiled, not run on hardware
//
#include <hip/hip_runtime.h>
#include <hip/hip_bf16.h>

// ---------------------------------------------------------------------------
// Performer attention for MI455X (gfx1250), wave32, bf16 WMMA everywhere.
// B=4 N=4096 D=1024 H=16 M=256 DH=64
// ---------------------------------------------------------------------------

typedef __bf16 bf16;
typedef __attribute__((ext_vector_type(16))) __bf16 bf16x16;
typedef __attribute__((ext_vector_type(8)))  __bf16 bf16x8;
typedef __attribute__((ext_vector_type(4)))  __bf16 bf16x4;
typedef __attribute__((ext_vector_type(2)))  __bf16 bf16x2;
typedef __attribute__((ext_vector_type(8)))  float  f32x8;
typedef __attribute__((ext_vector_type(4)))  float  f32x4;
typedef __attribute__((ext_vector_type(2)))  float  f32x2;

#define DEVINL static __device__ __forceinline__

// D = A(16x32 bf16) * B(32x16 bf16) + C(16x16 f32)
DEVINL f32x8 wmma_bf16(bf16x16 a, bf16x16 b, f32x8 c) {
  return __builtin_amdgcn_wmma_f32_16x16x32_bf16(
      false, a, false, b, (short)0, c, false, false);
}

// A fragment: tile stored row(k-major), row stride `rs` (elements).
// Lane L<16: row L, k {0..7,16..23}; lane L>=16: row L-16, k {8..15,24..31}.
DEVINL bf16x16 load_frag_a(const bf16* p, int rs, int kofs, int lane) {
  int r  = lane & 15;
  int hi = (lane >> 4) & 1;
  const bf16* q = p + r * rs + kofs;
  union { bf16x16 v; bf16x8 h[2]; } u;
  u.h[0] = *(const bf16x8*)(q + hi * 8);
  u.h[1] = *(const bf16x8*)(q + 16 + hi * 8);
  return u.v;
}

// B fragment: tile stored col(k-major) i.e. [n][k], row stride `rs`.
// Lane L<16: col L, k {0..7,16..23}; lane L>=16: col L-16, k {8..15,24..31}.
DEVINL bf16x16 load_frag_b(const bf16* p, int rs, int kofs, int lane) {
  int c  = lane & 15;
  int hi = (lane >> 4) & 1;
  const bf16* q = p + c * rs + kofs;
  union { bf16x16 v; bf16x8 h[2]; } u;
  u.h[0] = *(const bf16x8*)(q + hi * 8);
  u.h[1] = *(const bf16x8*)(q + 16 + hi * 8);
  return u.v;
}

// gfx1250 async global->LDS copy (16B per lane), tracked by ASYNCcnt.
DEVINL void async_load_b128(const void* gsrc, void* ldst) {
  unsigned           lds = (unsigned)(uintptr_t)ldst;     // low 32 bits = LDS offset
  unsigned long long ga  = (unsigned long long)(uintptr_t)gsrc;
  asm volatile("global_load_async_to_lds_b128 %0, %1, off"
               :: "v"(lds), "v"(ga) : "memory");
}
DEVINL void wait_async0() {
  asm volatile("s_wait_asynccnt 0" ::: "memory");
}

// ---------------------------------------------------------------------------
// Blocked GEMM: C[M,N] = A[M,K](f32) * B[K,N](f32) + bias[N], C in OutT.
// 256 threads (8 waves), tile 128x128, BK=32, DOUBLE-BUFFERED LDS:
// one barrier per K-step, next tile's global loads issued before compute.
// ---------------------------------------------------------------------------
template <typename OutT>
__global__ __launch_bounds__(256) void gemm_bias_kernel(
    const float* __restrict__ A, const float* __restrict__ B,
    const float* __restrict__ bias, OutT* __restrict__ C,
    int M, int N, int K) {
  __shared__ bf16 As[2][128 * 32];  // [m][k]
  __shared__ bf16 Bs[2][128 * 32];  // [n][k] (transposed)
  const int t    = threadIdx.x;
  const int lane = t & 31;
  const int w    = t >> 5;
  const int wy   = w >> 2;   // 0..1  -> m offset wy*64
  const int wx   = w & 3;    // 0..3  -> n offset wx*32
  const int mb   = blockIdx.x * 128;
  const int nb   = blockIdx.y * 128;

  f32x8 acc[4][2];
#pragma unroll
  for (int mi = 0; mi < 4; ++mi)
#pragma unroll
    for (int ni = 0; ni < 2; ++ni) acc[mi][ni] = (f32x8){};

  f32x4 ar[4];            // A staging regs
  f32x2 br0[4], br1[4];   // B staging regs (2x2 blocks)

  auto load_globals = [&](int kb) {
#pragma unroll
    for (int j = 0; j < 4; ++j) {          // A: 128x32, float4 chunks
      int c   = j * 256 + t;
      int row = c >> 3;
      int k4  = (c & 7) * 4;
      ar[j] = *(const f32x4*)(A + (size_t)(mb + row) * K + kb + k4);
    }
#pragma unroll
    for (int j = 0; j < 4; ++j) {          // B: 32x128, 2k x 2n blocks
      int c  = j * 256 + t;
      int kk = (c >> 6) * 2;
      int nn = (c & 63) * 2;
      const float* p = B + (size_t)(kb + kk) * N + nb + nn;
      br0[j] = *(const f32x2*)(p);
      br1[j] = *(const f32x2*)(p + N);
    }
  };
  auto store_lds = [&](int buf) {
#pragma unroll
    for (int j = 0; j < 4; ++j) {
      int c   = j * 256 + t;
      int row = c >> 3;
      int k4  = (c & 7) * 4;
      bf16x4 o = {(bf16)ar[j][0], (bf16)ar[j][1], (bf16)ar[j][2], (bf16)ar[j][3]};
      *(bf16x4*)(&As[buf][row * 32 + k4]) = o;            // one ds_store_b64
    }
#pragma unroll
    for (int j = 0; j < 4; ++j) {
      int c  = j * 256 + t;
      int kk = (c >> 6) * 2;
      int nn = (c & 63) * 2;
      bf16x2 c0 = {(bf16)br0[j][0], (bf16)br1[j][0]};     // col nn,  k kk..kk+1
      bf16x2 c1 = {(bf16)br0[j][1], (bf16)br1[j][1]};     // col nn+1
      *(bf16x2*)(&Bs[buf][(nn + 0) * 32 + kk]) = c0;      // ds_store_b32
      *(bf16x2*)(&Bs[buf][(nn + 1) * 32 + kk]) = c1;
    }
  };

  const int nk = K / 32;
  load_globals(0);
  store_lds(0);
  __syncthreads();

  for (int ik = 0; ik < nk; ++ik) {
    const int  cur     = ik & 1;
    const bool hasNext = (ik + 1) < nk;
    if (hasNext) load_globals((ik + 1) * 32);   // overlap with WMMA below

    bf16x16 bfrag[2];
#pragma unroll
    for (int ni = 0; ni < 2; ++ni)
      bfrag[ni] = load_frag_b(&Bs[cur][(wx * 32 + ni * 16) * 32], 32, 0, lane);
#pragma unroll
    for (int mi = 0; mi < 4; ++mi) {
      bf16x16 afrag = load_frag_a(&As[cur][(wy * 64 + mi * 16) * 32], 32, 0, lane);
#pragma unroll
      for (int ni = 0; ni < 2; ++ni)
        acc[mi][ni] = wmma_bf16(afrag, bfrag[ni], acc[mi][ni]);
    }

    if (hasNext) {
      store_lds(cur ^ 1);     // other buffer: no reader until after barrier
      __syncthreads();
    }
  }

  const int hi = lane >> 4, col = lane & 15;
#pragma unroll
  for (int mi = 0; mi < 4; ++mi)
#pragma unroll
    for (int ni = 0; ni < 2; ++ni) {
      int n    = nb + wx * 32 + ni * 16 + col;
      float bv = bias[n];
#pragma unroll
      for (int i = 0; i < 8; ++i) {
        int m = mb + wy * 64 + mi * 16 + i + 8 * hi;
        C[(size_t)m * N + n] = (OutT)(acc[mi][ni][i] + bv);
      }
    }
}

// ---------------------------------------------------------------------------
// Feature map: phi = exp(t @ omega - 0.5||t||^2).
// t tile staged via async global->LDS (pure bf16 copy); omega staged 2x2.
// ---------------------------------------------------------------------------
__global__ __launch_bounds__(256) void feature_map_kernel(
    const bf16* __restrict__ qkv, const float* __restrict__ omega,
    bf16* __restrict__ prime, float* __restrict__ ksum, int isK) {
  __shared__ bf16  Ts[128 * 64];   // t tile [n][dh]
  __shared__ bf16  Os[256 * 64];   // omega transposed [m][dh]
  __shared__ float npart[256];
  __shared__ float norms[128];
  const int t    = threadIdx.x;
  const int lane = t & 31;
  const int w    = t >> 5;
  const int nb   = blockIdx.x * 128;  // 32 chunks of N
  const int bh   = blockIdx.y;        // 64
  const int b    = bh >> 4, h = bh & 15;
  const int coff = isK * 1024 + h * 64;

  // async-stage t rows: 16B chunks, contiguous in both global and LDS
#pragma unroll
  for (int j = 0; j < 4; ++j) {
    int c = j * 256 + t;               // 1024 16B chunks
    int row = c >> 3, seg = (c & 7) * 8;
    async_load_b128(qkv + (size_t)(b * 4096 + nb + row) * 3072 + coff + seg,
                    Ts + row * 64 + seg);
  }
  // stage omega transposed ([dh][m] -> [m][dh]) as 2x2 blocks
#pragma unroll
  for (int j = 0; j < 16; ++j) {
    int c  = j * 256 + t;              // 4096 2x2 blocks
    int dh = (c >> 7) * 2;
    int m  = (c & 127) * 2;
    f32x2 r0 = *(const f32x2*)(omega + dh * 256 + m);
    f32x2 r1 = *(const f32x2*)(omega + (dh + 1) * 256 + m);
    bf16x2 w0 = {(bf16)r0[0], (bf16)r1[0]};
    bf16x2 w1 = {(bf16)r0[1], (bf16)r1[1]};
    *(bf16x2*)(&Os[(m + 0) * 64 + dh]) = w0;
    *(bf16x2*)(&Os[(m + 1) * 64 + dh]) = w1;
  }
  wait_async0();
  __syncthreads();

  // row norms: thread handles half a row
  {
    int row = t >> 1, half = t & 1;
    const bf16* p = Ts + row * 64 + half * 32;
    float s = 0.f;
#pragma unroll
    for (int e = 0; e < 32; ++e) { float x = (float)p[e]; s += x * x; }
    npart[t] = s;
  }
  __syncthreads();
  if (t < 128) norms[t] = 0.5f * (npart[2 * t] + npart[2 * t + 1]);
  __syncthreads();

  // wave w: rows w*16..w*16+15, all 16 m-tiles; K=64 => two wmma steps
  bf16x16 a0 = load_frag_a(Ts + w * 16 * 64, 64, 0, lane);
  bf16x16 a1 = load_frag_a(Ts + w * 16 * 64, 64, 32, lane);
  const int hi = lane >> 4, col = lane & 15;
  for (int mt = 0; mt < 16; ++mt) {
    bf16x16 b0 = load_frag_b(Os + mt * 16 * 64, 64, 0, lane);
    bf16x16 b1 = load_frag_b(Os + mt * 16 * 64, 64, 32, lane);
    f32x8 acc = (f32x8){};
    acc = wmma_bf16(a0, b0, acc);
    acc = wmma_bf16(a1, b1, acc);
    float csum = 0.f;
#pragma unroll
    for (int i = 0; i < 8; ++i) {
      int r = w * 16 + i + 8 * hi;
      float phi = __expf(acc[i] - norms[r]);
      csum += phi;
      prime[(size_t)(bh * 4096 + nb + r) * 256 + mt * 16 + col] = (bf16)phi;
    }
    if (isK) atomicAdd(ksum + bh * 256 + mt * 16 + col, csum);
  }
}

// ---------------------------------------------------------------------------
// k_v[b,h,m,dh] = sum_n k'[n,m] * v[n,dh].  Split-K: grid (bh, 8 chunks of 512).
// ---------------------------------------------------------------------------
__global__ __launch_bounds__(256) void kv_kernel(
    const bf16* __restrict__ kp, const bf16* __restrict__ qkv,
    float* __restrict__ kv) {
  __shared__ bf16 kps[256 * 32];  // [m][n]
  __shared__ bf16 vs[64 * 32];    // [dh][n]
  const int t    = threadIdx.x;
  const int lane = t & 31;
  const int w    = t >> 5;
  const int bh   = blockIdx.x;    // 64
  const int b    = bh >> 4, h = bh & 15;
  const int nb0  = blockIdx.y * 512;

  f32x8 acc[2][4];
#pragma unroll
  for (int mi = 0; mi < 2; ++mi)
#pragma unroll
    for (int dt = 0; dt < 4; ++dt) acc[mi][dt] = (f32x8){};

  for (int ks = 0; ks < 16; ++ks) {
    const int nb = nb0 + ks * 32;
    // k' tile [32n][256m] -> kps[m][n], 2n x 2m blocks
#pragma unroll
    for (int j = 0; j < 8; ++j) {
      int c = j * 256 + t;             // 2048 blocks
      int n = (c >> 7) * 2;
      int m = (c & 127) * 2;
      const bf16* src = kp + (size_t)(bh * 4096 + nb + n) * 256 + m;
      bf16x2 r0 = *(const bf16x2*)(src);
      bf16x2 r1 = *(const bf16x2*)(src + 256);
      bf16x2 w0 = {r0[0], r1[0]};
      bf16x2 w1 = {r0[1], r1[1]};
      *(bf16x2*)(&kps[(m + 0) * 32 + n]) = w0;
      *(bf16x2*)(&kps[(m + 1) * 32 + n]) = w1;
    }
    // v tile [32n][64dh] -> vs[dh][n]
#pragma unroll
    for (int j = 0; j < 8; ++j) {
      int c = j * 256 + t;             // 2048 elements
      int n = c >> 6, dh = c & 63;
      vs[dh * 32 + n] =
          qkv[(size_t)(b * 4096 + nb + n) * 3072 + 2048 + h * 64 + dh];
    }
    __syncthreads();

    bf16x16 bfrag[4];
#pragma unroll
    for (int dt = 0; dt < 4; ++dt)
      bfrag[dt] = load_frag_b(vs + dt * 16 * 32, 32, 0, lane);
#pragma unroll
    for (int mi = 0; mi < 2; ++mi) {
      bf16x16 afrag = load_frag_a(kps + (w * 2 + mi) * 16 * 32, 32, 0, lane);
#pragma unroll
      for (int dt = 0; dt < 4; ++dt)
        acc[mi][dt] = wmma_bf16(afrag, bfrag[dt], acc[mi][dt]);
    }
    __syncthreads();
  }

  const int hi = lane >> 4, col = lane & 15;
#pragma unroll
  for (int mi = 0; mi < 2; ++mi)
#pragma unroll
    for (int dt = 0; dt < 4; ++dt)
#pragma unroll
      for (int i = 0; i < 8; ++i) {
        int m  = (w * 2 + mi) * 16 + i + 8 * hi;
        int dh = dt * 16 + col;
        atomicAdd(kv + (size_t)bh * 16384 + m * 64 + dh, acc[mi][dt][i]);
      }
}

// ---------------------------------------------------------------------------
// numerator = q' @ k_v ; denominator = q' . k_sum ; out = num/(den+1e-6)
// ---------------------------------------------------------------------------
__global__ __launch_bounds__(256) void attn_combine_kernel(
    const bf16* __restrict__ qp, const float* __restrict__ kv,
    const float* __restrict__ ksum, float* __restrict__ out) {
  __shared__ bf16  kvs[64 * 256];   // [dh][m]
  __shared__ float ksums[256];
  __shared__ float dens[128];
  const int t    = threadIdx.x;
  const int lane = t & 31;
  const int w    = t >> 5;
  const int bh   = blockIdx.x;
  const int b    = bh >> 4, h = bh & 15;
  const int nb   = blockIdx.y * 128;

  // stage k_v transposed (f32 -> bf16), 2m x 2dh blocks
#pragma unroll
  for (int j = 0; j < 16; ++j) {
    int c  = j * 256 + t;            // 4096 blocks
    int m  = (c >> 5) * 2;
    int dh = (c & 31) * 2;
    const float* src = kv + (size_t)bh * 16384 + m * 64 + dh;
    f32x2 r0 = *(const f32x2*)(src);
    f32x2 r1 = *(const f32x2*)(src + 64);
    bf16x2 w0 = {(bf16)r0[0], (bf16)r1[0]};
    bf16x2 w1 = {(bf16)r0[1], (bf16)r1[1]};
    *(bf16x2*)(&kvs[(dh + 0) * 256 + m]) = w0;
    *(bf16x2*)(&kvs[(dh + 1) * 256 + m]) = w1;
  }
  ksums[t] = ksum[bh * 256 + t];
  __syncthreads();

  const int hi = lane >> 4, col = lane & 15;
  const bf16* qrowbase = qp + (size_t)(bh * 4096 + nb + w * 16) * 256;

  f32x8 acc[4];
#pragma unroll
  for (int dt = 0; dt < 4; ++dt) acc[dt] = (f32x8){};

#pragma unroll
  for (int ksb = 0; ksb < 8; ++ksb) {
    int kofs = ksb * 32;
    bf16x16 afrag = load_frag_a(qrowbase, 256, kofs, lane);  // global b128 loads
#pragma unroll
    for (int dt = 0; dt < 4; ++dt) {
      bf16x16 bfrag = load_frag_b(kvs + dt * 16 * 256, 256, kofs, lane);
      acc[dt] = wmma_bf16(afrag, bfrag, acc[dt]);
    }
  }

  // denominator: each lane does half a row, halves combined via shuffle
  {
    int r = lane & 15;
    const bf16* qrow = qrowbase + (size_t)r * 256 + hi * 128;
    float den = 0.f;
#pragma unroll
    for (int c8 = 0; c8 < 16; ++c8) {
      bf16x8 v = *(const bf16x8*)(qrow + c8 * 8);
#pragma unroll
      for (int e = 0; e < 8; ++e)
        den += (float)v[e] * ksums[hi * 128 + c8 * 8 + e];
    }
    den += __shfl_down(den, 16);
    if (hi == 0) dens[w * 16 + r] = den + 1e-6f;
  }
  __syncthreads();

#pragma unroll
  for (int dt = 0; dt < 4; ++dt)
#pragma unroll
    for (int i = 0; i < 8; ++i) {
      int r = w * 16 + i + 8 * hi;
      out[(size_t)(b * 4096 + nb + r) * 1024 + h * 64 + dt * 16 + col] =
          acc[dt][i] / dens[r];
    }
}

__global__ void zero_kernel(float* __restrict__ p, size_t n) {
  size_t i = (size_t)blockIdx.x * blockDim.x + threadIdx.x;
  size_t s = (size_t)gridDim.x * blockDim.x;
  for (; i < n; i += s) p[i] = 0.f;
}

// ---------------------------------------------------------------------------
// Workspace layout (bytes):
//   qkv (bf16)  [16384,3072] :   0        .. 100663296
//   q'  (bf16)  [64,4096,256]: 100663296 .. 134217728
//   k'  (bf16)  [64,4096,256]: 134217728 .. 167772160
//   ksum (f32)  [64,256]     : 167772160 .. 167837696
//   k_v  (f32)  [64,256,64]  : 167837696 .. 172032000
//   attn (f32)  [16384,1024] : 172032000 .. 239140864   (~228 MB total)
// ---------------------------------------------------------------------------
extern "C" void kernel_launch(void* const* d_in, const int* in_sizes, int n_in,
                              void* d_out, int out_size, void* d_ws, size_t ws_size,
                              hipStream_t stream) {
  (void)in_sizes; (void)n_in; (void)out_size; (void)ws_size;
  const float* x     = (const float*)d_in[0];
  const float* Wqkv  = (const float*)d_in[1];
  const float* bqkv  = (const float*)d_in[2];
  const float* Wout  = (const float*)d_in[3];
  const float* bout  = (const float*)d_in[4];
  const float* omega = (const float*)d_in[5];
  float* outp = (float*)d_out;

  char* ws = (char*)d_ws;
  bf16*  qkv_bf = (bf16*)(ws + 0);
  bf16*  qp     = (bf16*)(ws + 100663296ull);
  bf16*  kp     = (bf16*)(ws + 134217728ull);
  float* ksum   = (float*)(ws + 167772160ull);
  float* kvbuf  = (float*)(ws + 167837696ull);
  float* attn   = (float*)(ws + 172032000ull);

  // 1) qkv = x @ Wqkv + bqkv  (store bf16)
  gemm_bias_kernel<bf16><<<dim3(128, 24), 256, 0, stream>>>(
      x, Wqkv, bqkv, qkv_bf, 16384, 3072, 1024);

  // 2) zero ksum (16K f32) + k_v (1M f32), contiguous range
  zero_kernel<<<256, 256, 0, stream>>>(ksum, (65536ull + 4194304ull) / 4);

  // 3) q' = exp(q@omega - 0.5||q||^2)
  feature_map_kernel<<<dim3(32, 64), 256, 0, stream>>>(qkv_bf, omega, qp, ksum, 0);
  // 4) k' = exp(k@omega - 0.5||k||^2), accumulate k_sum
  feature_map_kernel<<<dim3(32, 64), 256, 0, stream>>>(qkv_bf, omega, kp, ksum, 1);

  // 5) k_v = k'^T @ v  (split-K, f32 atomics)
  kv_kernel<<<dim3(64, 8), 256, 0, stream>>>(kp, qkv_bf, kvbuf);

  // 6) attn = (q' @ k_v) / (q' . k_sum + 1e-6), [b,n,d] layout
  attn_combine_kernel<<<dim3(64, 32), 256, 0, stream>>>(qp, kvbuf, ksum, attn);

  // 7) out = attn @ Wout + bout
  gemm_bias_kernel<float><<<dim3(128, 8), 256, 0, stream>>>(
      attn, Wout, bout, outp, 16384, 1024, 1024);
}
